// Node_align_Node_loss_34505767256122
// MI455X (gfx1250) — compile-verified, matrix operand order checked by reading, not run on hardware
//
#include <hip/hip_runtime.h>
#include <hip/hip_bf16.h>

typedef __bf16 bf16;
typedef __attribute__((ext_vector_type(16))) __bf16 v16bf;
typedef __attribute__((ext_vector_type(8)))  float  v8f;
typedef __attribute__((ext_vector_type(4)))  float  f32v4;
typedef __attribute__((ext_vector_type(4)))  unsigned int u32v4;
typedef __attribute__((ext_vector_type(2)))  unsigned int u32v2;
typedef __attribute__((ext_vector_type(8)))  int i32v8;
typedef __attribute__((ext_vector_type(4)))  int i32v4;

#define TEMP_INV 10.0f   // 1/0.1

#if __has_builtin(__builtin_amdgcn_tensor_load_to_lds) && \
    __has_builtin(__builtin_amdgcn_s_wait_tensorcnt)
#define USE_TDM 1
#else
#define USE_TDM 0
#endif

// ---------- helpers ----------
__device__ __forceinline__ bf16 f2bf(float f) {
  union { float f; unsigned u; } in; in.f = f;
  unsigned r = (in.u + 0x7FFFu + ((in.u >> 16) & 1u)) >> 16;
  union { unsigned short s; bf16 b; } out; out.s = (unsigned short)r;
  return out.b;
}
__device__ __forceinline__ unsigned pk2(float a, float b) {
  union { unsigned short s[2]; unsigned u; } o;
  union { bf16 b; unsigned short s; } x, y;
  x.b = f2bf(a); y.b = f2bf(b);
  o.s[0] = x.s; o.s[1] = y.s;
  return o.u;
}

__device__ __forceinline__ v8f wmma_bf(v16bf a, v16bf b, v8f c) {
  // D = A(16x32 bf16) * B(32x16 bf16) + C(16x16 f32)
  return __builtin_amdgcn_wmma_f32_16x16x32_bf16(false, a, false, b, (short)0, c,
                                                 false, false);
}

// K-element index inside a 32-wide K chunk for 16-bit A/B fragments
// (ISA 7.12.2: VGPR0-3 hold K 0..7 / 8..15 by lane half, VGPR4-7 hold +16)
__device__ __forceinline__ int kidx(int i, int kh) {
  return ((i & 8) << 1) + (kh << 3) + (i & 7);
}

__device__ __forceinline__ v16bf ldsA(const bf16* base, int ld, int m, int kh, int kb) {
  v16bf a;
#pragma unroll
  for (int i = 0; i < 16; ++i) a[i] = base[m * ld + kb + kidx(i, kh)];
  return a;
}
__device__ __forceinline__ v16bf gmemB(const bf16* __restrict__ W, int ldn, int n, int kh, int kb) {
  v16bf b;
#pragma unroll
  for (int i = 0; i < 16; ++i) b[i] = W[(size_t)(kb + kidx(i, kh)) * ldn + n];
  return b;
}
__device__ __forceinline__ v16bf ldsB(const bf16* M, int ldn, int n, int kh, int kb) {
  v16bf b;
#pragma unroll
  for (int i = 0; i < 16; ++i) b[i] = M[(kb + kidx(i, kh)) * ldn + n];
  return b;
}
// B fragment for A*B^T: element (k,n) = Mt[n*ld + k]
__device__ __forceinline__ v16bf ldsBT(const bf16* Mt, int ld, int n, int kh, int kb) {
  v16bf b;
#pragma unroll
  for (int i = 0; i < 16; ++i) b[i] = Mt[n * ld + kb + kidx(i, kh)];
  return b;
}

// ---------- TDM: 2D 128x128 bf16 tile, global -> LDS (ISA ch.8 D# layout) ----------
__device__ __forceinline__ void tdm_load_128x128_bf16(const bf16* gsrc, void* ldst) {
#if USE_TDM
  unsigned long long ga = (unsigned long long)gsrc;
  u32v4 g0;
  g0[0] = 1u;                                          // count=1, user mode
  g0[1] = (unsigned)(unsigned long long)ldst;          // lds_addr (bytes, addr[31:0])
  g0[2] = (unsigned)ga;                                // global_addr[31:0]
  g0[3] = ((unsigned)(ga >> 32) & 0x01FFFFFFu) | 0x80000000u;  // addr[56:32] | type=2
  i32v8 g1;
  g1[0] = 0x00010000;          // workgroup_mask=0, data_size=1 (2 bytes)
  g1[1] = (int)(128u << 16);   // atomic_bar=0 | tensor_dim0 lo16 = 128
  g1[2] = (int)(128u << 16);   // tensor_dim0 hi=0 | tensor_dim1 lo16 = 128
  g1[3] = (int)(128u << 16);   // tensor_dim1 hi=0 | tile_dim0 = 128
  g1[4] = 128;                 // tile_dim1=128 | tile_dim2=0
  g1[5] = 128;                 // tensor_dim0_stride lo32 = 128
  g1[6] = (int)(16384u << 16); // dim0_stride hi=0 | tensor_dim1_stride lo16
  g1[7] = 0;                   // tensor_dim1_stride hi
  i32v4 z4 = {0, 0, 0, 0};     // groups 2/3 unused (2D)
  i32v8 z8 = {0, 0, 0, 0, 0, 0, 0, 0};
  // clang-23 / therock form: 6 args (g0, g1, g2, g3, extra, cpol)
  __builtin_amdgcn_tensor_load_to_lds(g0, g1, z4, z4, z8, 0);
#else
  (void)gsrc; (void)ldst;
#endif
}

// ---------- weight conversion ----------
__global__ void k_f2bf(const float* __restrict__ s, bf16* __restrict__ d, int n) {
  int i = blockIdx.x * 256 + threadIdx.x;
  if (i < n) d[i] = f2bf(s[i]);
}

// ---------- prefix scan of set sizes -> starts[1024] ----------
__global__ void k_starts(const int* __restrict__ qs, const int* __restrict__ cs,
                         int* __restrict__ starts) {
  __shared__ int s[1024];
  int t = threadIdx.x;
  int sz = (t & 1) ? cs[t >> 1] : qs[t >> 1];
  s[t] = sz;
  __syncthreads();
  for (int off = 1; off < 1024; off <<= 1) {
    int v = (t >= off) ? s[t - off] : 0;
    __syncthreads();
    s[t] += v;
    __syncthreads();
  }
  starts[t] = s[t] - sz;  // exclusive scan
}

// ---------- node encoder: h = x(Nx32) @ W(32x128) + b ----------
__global__ __launch_bounds__(256)
void k_enc_nodes(const float* __restrict__ x, const bf16* __restrict__ W,
                 const float* __restrict__ bias, float* __restrict__ h,
                 bf16* __restrict__ hbf, int N) {
  __shared__ bf16 sA[16 * 32];
  int row0 = blockIdx.x * 16, tid = threadIdx.x;
  for (int i = tid; i < 16 * 32; i += 256) {
    int r = i >> 5, c = i & 31;
    float v = (row0 + r < N) ? x[(size_t)(row0 + r) * 32 + c] : 0.f;
    sA[i] = f2bf(v);
  }
  __syncthreads();
  int wave = tid >> 5, lane = tid & 31, nn = lane & 15, kh = (lane >> 4) & 1, m = lane & 15;
  int col = wave * 16 + nn;
  v8f acc = {};
  acc = wmma_bf(ldsA(sA, 32, m, kh, 0), gmemB(W, 128, col, kh, 0), acc);
#pragma unroll
  for (int r = 0; r < 8; ++r) {
    int mm = r + (kh << 3);
    int row = row0 + mm;
    if (row < N) {
      float v = acc[r] + bias[col];
      h[(size_t)row * 128 + col] = v;
      hbf[(size_t)row * 128 + col] = f2bf(v);
    }
  }
}

// ---------- edge encoder: e = x(Ex16) @ W(16x64) + b (K padded to 32) ----------
__global__ __launch_bounds__(128)
void k_enc_edges(const float* __restrict__ x, const bf16* __restrict__ W,
                 const float* __restrict__ bias, bf16* __restrict__ ebf, int E) {
  __shared__ bf16 sA[16 * 32];
  int row0 = blockIdx.x * 16, tid = threadIdx.x;
  for (int i = tid; i < 16 * 32; i += 128) {
    int r = i >> 5, c = i & 31;
    float v = (c < 16 && row0 + r < E) ? x[(size_t)(row0 + r) * 16 + c] : 0.f;
    sA[i] = f2bf(v);
  }
  __syncthreads();
  int wave = tid >> 5, lane = tid & 31, nn = lane & 15, kh = (lane >> 4) & 1, m = lane & 15;
  int col = wave * 16 + nn;  // 4 waves -> 64 cols
  v16bf bfr;
#pragma unroll
  for (int i = 0; i < 16; ++i) {
    int k = kidx(i, kh);
    bfr[i] = (k < 16) ? W[k * 64 + col] : f2bf(0.f);
  }
  v8f acc = {};
  acc = wmma_bf(ldsA(sA, 32, m, kh, 0), bfr, acc);
#pragma unroll
  for (int r = 0; r < 8; ++r) {
    int mm = r + (kh << 3);
    int row = row0 + mm;
    if (row < E) ebf[(size_t)row * 64 + col] = f2bf(acc[r] + bias[col]);
  }
}

// ---------- per-edge message MLPs (both directions) + segment-sum atomics ----------
__global__ __launch_bounds__(256)
void k_msg(const bf16* __restrict__ hbf, const bf16* __restrict__ ebf,
           const int* __restrict__ fidx, const int* __restrict__ tidx,
           const bf16* __restrict__ Wm1, const float* __restrict__ bm1,
           const bf16* __restrict__ Wm2, const float* __restrict__ bm2,
           const bf16* __restrict__ Wr1, const float* __restrict__ br1,
           const bf16* __restrict__ Wr2, const float* __restrict__ br2,
           float* __restrict__ agg, int E) {
  __shared__ bf16 sX[16 * 320];  // [src(128) | dst(128) | e(64)]
  __shared__ bf16 sH[16 * 256];
  int e0 = blockIdx.x * 16, tid = threadIdx.x;
  // gather & stage activations: 16-byte vector path (rows of h are 256B contiguous)
  {
    int r = tid >> 4, c4 = tid & 15;  // 256 threads -> 16 rows x 16 u32v4
    int e = e0 + r;
    u32v4 zs = {0, 0, 0, 0}, zd = {0, 0, 0, 0};
    if (e < E) {
      zs = ((const u32v4*)(hbf + (size_t)fidx[e] * 128))[c4];
      zd = ((const u32v4*)(hbf + (size_t)tidx[e] * 128))[c4];
    }
    ((u32v4*)(sX + r * 320))[c4] = zs;
    ((u32v4*)(sX + r * 320 + 128))[c4] = zd;
  }
  for (int i = tid; i < 16 * 8; i += 256) {  // e: 16 rows x 8 u32v4
    int r = i >> 3, c4 = i & 7;
    int e = e0 + r;
    u32v4 ve = {0, 0, 0, 0};
    if (e < E) ve = ((const u32v4*)(ebf + (size_t)e * 64))[c4];
    ((u32v4*)(sX + r * 320 + 256))[c4] = ve;
  }
  __syncthreads();

  int wave = tid >> 5, lane = tid & 31, nn = lane & 15, kh = (lane >> 4) & 1, m = lane & 15;
  for (int d = 0; d < 2; ++d) {
    const bf16* W1 = d ? Wr1 : Wm1;  const float* B1 = d ? br1 : bm1;
    const bf16* W2 = d ? Wr2 : Wm2;  const float* B2 = d ? br2 : bm2;
    // GEMM1: (16x320) x (320x256), ReLU -> sH.  Reverse dir swaps src/dst via k^128.
    for (int t = 0; t < 2; ++t) {
      int col = (wave + 8 * t) * 16 + nn;
      v8f acc = {};
      for (int kb = 0; kb < 320; kb += 32) {
        v16bf a;
#pragma unroll
        for (int i = 0; i < 16; ++i) {
          int k = kb + kidx(i, kh);
          int ks = (d && k < 256) ? (k ^ 128) : k;
          a[i] = sX[m * 320 + ks];
        }
        acc = wmma_bf(a, gmemB(W1, 256, col, kh, kb), acc);
      }
#pragma unroll
      for (int r = 0; r < 8; ++r) {
        int mm = r + (kh << 3);
        sH[mm * 256 + col] = f2bf(fmaxf(acc[r] + B1[col], 0.f));
      }
    }
    __syncthreads();
    // GEMM2: (16x256) x (256x128) + atomic scatter-add into agg
    {
      int col = wave * 16 + nn;
      v8f acc = {};
      for (int kb = 0; kb < 256; kb += 32)
        acc = wmma_bf(ldsA(sH, 256, m, kh, kb), gmemB(W2, 128, col, kh, kb), acc);
#pragma unroll
      for (int r = 0; r < 8; ++r) {
        int mm = r + (kh << 3);
        int e = e0 + mm;
        if (e < E) {
          int tgt = d ? fidx[e] : tidx[e];  // fwd -> to_idx, rev -> from_idx
          atomicAdd(&agg[(size_t)tgt * 128 + col], acc[r] + B2[col]);
        }
      }
    }
    __syncthreads();
  }
}

// ---------- node update: h += MLP([agg|h]) ----------
__global__ __launch_bounds__(256)
void k_upd(const float* __restrict__ agg, float* __restrict__ h, bf16* __restrict__ hbf,
           const bf16* __restrict__ W1, const float* __restrict__ B1,
           const bf16* __restrict__ W2, const float* __restrict__ B2, int N) {
  __shared__ bf16 sX[16 * 256];
  __shared__ bf16 sH[16 * 256];
  int n0 = blockIdx.x * 16, tid = threadIdx.x;
  for (int i = tid; i < 16 * 32; i += 256) {  // agg: f32v4 load -> packed bf16 store
    int r = i >> 5, c4 = i & 31;
    int n = n0 + r;
    f32v4 v = {0.f, 0.f, 0.f, 0.f};
    if (n < N) v = ((const f32v4*)(agg + (size_t)n * 128))[c4];
    u32v2 o; o[0] = pk2(v.x, v.y); o[1] = pk2(v.z, v.w);
    ((u32v2*)(sX + r * 256))[c4] = o;
  }
  {
    int r = tid >> 4, c4 = tid & 15;  // h: 16B copies
    int n = n0 + r;
    u32v4 v = {0, 0, 0, 0};
    if (n < N) v = ((const u32v4*)(hbf + (size_t)n * 128))[c4];
    ((u32v4*)(sX + r * 256 + 128))[c4] = v;
  }
  __syncthreads();
  int wave = tid >> 5, lane = tid & 31, nn = lane & 15, kh = (lane >> 4) & 1, m = lane & 15;
  for (int t = 0; t < 2; ++t) {
    int col = (wave + 8 * t) * 16 + nn;
    v8f acc = {};
    for (int kb = 0; kb < 256; kb += 32)
      acc = wmma_bf(ldsA(sX, 256, m, kh, kb), gmemB(W1, 256, col, kh, kb), acc);
#pragma unroll
    for (int r = 0; r < 8; ++r) {
      int mm = r + (kh << 3);
      sH[mm * 256 + col] = f2bf(fmaxf(acc[r] + B1[col], 0.f));
    }
  }
  __syncthreads();
  {
    int col = wave * 16 + nn;
    v8f acc = {};
    for (int kb = 0; kb < 256; kb += 32)
      acc = wmma_bf(ldsA(sH, 256, m, kh, kb), gmemB(W2, 128, col, kh, kb), acc);
#pragma unroll
    for (int r = 0; r < 8; ++r) {
      int mm = r + (kh << 3);
      int n = n0 + mm;
      if (n < N) {
        float v = h[(size_t)n * 128 + col] + acc[r] + B2[col];
        h[(size_t)n * 128 + col] = v;
        hbf[(size_t)n * 128 + col] = f2bf(v);
      }
    }
  }
}

// ---------- scatter h into (B*2, 128, 128) padded sets (8 nodes/block) ----------
__global__ __launch_bounds__(256)
void k_scatter(const float* __restrict__ h, const int* __restrict__ gidx,
               const int* __restrict__ starts, float* __restrict__ stk,
               bf16* __restrict__ stkbf, int N) {
  int tid = threadIdx.x;
  int n = blockIdx.x * 8 + (tid >> 5), c4 = tid & 31;
  if (n >= N) return;
  int g = gidx[n];
  int pos = n - starts[g];
  if (pos < 0 || pos >= 128) return;
  f32v4 v = ((const f32v4*)(h + (size_t)n * 128))[c4];
  size_t o = (size_t)g * 16384 + (size_t)pos * 128;
  ((f32v4*)(stk + o))[c4] = v;
  u32v2 p; p[0] = pk2(v.x, v.y); p[1] = pk2(v.z, v.w);
  ((u32v2*)(stkbf + o))[c4] = p;
}

// ---------- per-set transform MLP + mask (tq/tc) ----------
__global__ __launch_bounds__(256)
void k_tqc(const bf16* __restrict__ stkbf, const bf16* __restrict__ Wf1,
           const float* __restrict__ bf1, const bf16* __restrict__ Wf2,
           const float* __restrict__ bf2, const int* __restrict__ qs,
           const int* __restrict__ cs, bf16* __restrict__ mqc) {
  __shared__ bf16 sIn[128 * 128];
  __shared__ bf16 sHid[128 * 128];
  int s = blockIdx.x, tid = threadIdx.x;
  int pair = s >> 1, side = s & 1;
  int size = side ? cs[pair] : qs[pair];
  const bf16* in = stkbf + (size_t)s * 16384;
#if USE_TDM
  if (tid < 32) {
    tdm_load_128x128_bf16(in, sIn);
    __builtin_amdgcn_s_wait_tensorcnt(0);
  }
#else
  for (int i = tid; i < 2048; i += 256) ((u32v4*)sIn)[i] = ((const u32v4*)in)[i];
#endif
  __syncthreads();
  int wave = tid >> 5, lane = tid & 31, nn = lane & 15, kh = (lane >> 4) & 1, m = lane & 15;
  int rt = wave;
  for (int ct = 0; ct < 8; ++ct) {
    int col = ct * 16 + nn;
    v8f acc = {};
    for (int kb = 0; kb < 128; kb += 32)
      acc = wmma_bf(ldsA(sIn, 128, rt * 16 + m, kh, kb), gmemB(Wf1, 128, col, kh, kb), acc);
#pragma unroll
    for (int r = 0; r < 8; ++r) {
      int mm = rt * 16 + r + (kh << 3);
      sHid[mm * 128 + col] = f2bf(fmaxf(acc[r] + bf1[col], 0.f));
    }
  }
  __syncthreads();
  bf16* outp = mqc + (size_t)s * 16384;
  for (int ct = 0; ct < 8; ++ct) {
    int col = ct * 16 + nn;
    v8f acc = {};
    for (int kb = 0; kb < 128; kb += 32)
      acc = wmma_bf(ldsA(sHid, 128, rt * 16 + m, kh, kb), gmemB(Wf2, 128, col, kh, kb), acc);
#pragma unroll
    for (int r = 0; r < 8; ++r) {
      int mm = rt * 16 + r + (kh << 3);
      float v = (mm < size) ? (acc[r] + bf2[col]) : 0.f;
      outp[mm * 128 + col] = f2bf(v);
    }
  }
}

// ---------- log_alpha = (mq @ mc^T) / TEMP ----------
__global__ __launch_bounds__(256)
void k_logalpha(const bf16* __restrict__ mqc, float* __restrict__ la) {
  __shared__ bf16 sQ[128 * 128];
  __shared__ bf16 sC[128 * 128];
  int p = blockIdx.x, tid = threadIdx.x;
  const bf16* q = mqc + (size_t)(2 * p) * 16384;
  const bf16* c = mqc + (size_t)(2 * p + 1) * 16384;
#if USE_TDM
  if (tid < 32) {
    tdm_load_128x128_bf16(q, sQ);
    tdm_load_128x128_bf16(c, sC);
    __builtin_amdgcn_s_wait_tensorcnt(0);
  }
#else
  for (int i = tid; i < 2048; i += 256) {
    ((u32v4*)sQ)[i] = ((const u32v4*)q)[i];
    ((u32v4*)sC)[i] = ((const u32v4*)c)[i];
  }
#endif
  __syncthreads();
  int wave = tid >> 5, lane = tid & 31, nn = lane & 15, kh = (lane >> 4) & 1, m = lane & 15;
  int rt = wave;
  float* outp = la + (size_t)p * 16384;
  for (int ct = 0; ct < 8; ++ct) {
    int col = ct * 16 + nn;
    v8f acc = {};
    for (int kb = 0; kb < 128; kb += 32)
      acc = wmma_bf(ldsA(sQ, 128, rt * 16 + m, kh, kb), ldsBT(sC, 128, col, kh, kb), acc);
#pragma unroll
    for (int r = 0; r < 8; ++r) {
      int mm = rt * 16 + r + (kh << 3);
      outp[mm * 128 + col] = acc[r] * TEMP_INV;
    }
  }
}

// ---------- Sinkhorn (20 iters) fully in LDS, then plan = exp ----------
__global__ __launch_bounds__(256)
void k_sinkhorn(const float* __restrict__ la, bf16* __restrict__ planbf) {
  __shared__ float sLA[128 * 129];  // +1 pad: conflict-free column access
  __shared__ float sR[128];
  int p = blockIdx.x, tid = threadIdx.x;
  const float* in = la + (size_t)p * 16384;
  for (int i = tid; i < 16384; i += 256) {
    int r = i >> 7, c = i & 127;
    sLA[r * 129 + c] = in[i];
  }
  __syncthreads();
  for (int it = 0; it < 20; ++it) {
    if (tid < 128) {  // axis=2 (rows)
      float mx = -3.0e38f;
      for (int c = 0; c < 128; ++c) mx = fmaxf(mx, sLA[tid * 129 + c]);
      float sum = 0.f;
      for (int c = 0; c < 128; ++c) sum += __expf(sLA[tid * 129 + c] - mx);
      sR[tid] = mx + __logf(sum);
    }
    __syncthreads();
    for (int i = tid; i < 16384; i += 256) {
      int r = i >> 7, c = i & 127;
      sLA[r * 129 + c] -= sR[r];
    }
    __syncthreads();
    if (tid < 128) {  // axis=1 (cols)
      float mx = -3.0e38f;
      for (int r = 0; r < 128; ++r) mx = fmaxf(mx, sLA[r * 129 + tid]);
      float sum = 0.f;
      for (int r = 0; r < 128; ++r) sum += __expf(sLA[r * 129 + tid] - mx);
      sR[tid] = mx + __logf(sum);
    }
    __syncthreads();
    for (int i = tid; i < 16384; i += 256) {
      int r = i >> 7, c = i & 127;
      sLA[r * 129 + c] -= sR[c];
    }
    __syncthreads();
  }
  bf16* outp = planbf + (size_t)p * 16384;
  for (int i = tid; i < 16384; i += 256) {
    int r = i >> 7, c = i & 127;
    outp[i] = f2bf(__expf(sLA[r * 129 + c]));
  }
}

// ---------- scores = -sum relu(stacked_q - plan @ stacked_c) ----------
__global__ __launch_bounds__(256)
void k_scores(const bf16* __restrict__ planbf, const bf16* __restrict__ stkbf,
              const float* __restrict__ stkf, float* __restrict__ out) {
  __shared__ bf16 sP[128 * 128];
  __shared__ bf16 sC[128 * 128];
  __shared__ float red[256];
  int p = blockIdx.x, tid = threadIdx.x;
  const bf16* pl = planbf + (size_t)p * 16384;
  const bf16* c = stkbf + (size_t)(2 * p + 1) * 16384;
  const float* qf = stkf + (size_t)(2 * p) * 16384;
#if USE_TDM
  if (tid < 32) {
    tdm_load_128x128_bf16(pl, sP);
    tdm_load_128x128_bf16(c, sC);
    __builtin_amdgcn_s_wait_tensorcnt(0);
  }
#else
  for (int i = tid; i < 2048; i += 256) {
    ((u32v4*)sP)[i] = ((const u32v4*)pl)[i];
    ((u32v4*)sC)[i] = ((const u32v4*)c)[i];
  }
#endif
  __syncthreads();
  int wave = tid >> 5, lane = tid & 31, nn = lane & 15, kh = (lane >> 4) & 1, m = lane & 15;
  int rt = wave;
  float partial = 0.f;
  for (int ct = 0; ct < 8; ++ct) {
    int col = ct * 16 + nn;
    v8f acc = {};
    for (int kb = 0; kb < 128; kb += 32)
      acc = wmma_bf(ldsA(sP, 128, rt * 16 + m, kh, kb), ldsB(sC, 128, col, kh, kb), acc);
#pragma unroll
    for (int r = 0; r < 8; ++r) {
      int q = rt * 16 + r + (kh << 3);
      float v = qf[q * 128 + col] - acc[r];
      partial += fmaxf(v, 0.f);
    }
  }
  red[tid] = partial;
  __syncthreads();
  for (int s2 = 128; s2 > 0; s2 >>= 1) {
    if (tid < s2) red[tid] += red[tid + s2];
    __syncthreads();
  }
  if (tid == 0) out[p] = -red[0];
}

// ---------- host driver ----------
extern "C" void kernel_launch(void* const* d_in, const int* in_sizes, int n_in,
                              void* d_out, int out_size, void* d_ws, size_t ws_size,
                              hipStream_t stream) {
  const float* node_f      = (const float*)d_in[0];
  const float* edge_f      = (const float*)d_in[1];
  const float* enc_node_w  = (const float*)d_in[2];
  const float* enc_node_b  = (const float*)d_in[3];
  const float* enc_edge_w  = (const float*)d_in[4];
  const float* enc_edge_b  = (const float*)d_in[5];
  const float* msg_w1      = (const float*)d_in[6];
  const float* msg_b1      = (const float*)d_in[7];
  const float* msg_w2      = (const float*)d_in[8];
  const float* msg_b2      = (const float*)d_in[9];
  const float* rmsg_w1     = (const float*)d_in[10];
  const float* rmsg_b1     = (const float*)d_in[11];
  const float* rmsg_w2     = (const float*)d_in[12];
  const float* rmsg_b2     = (const float*)d_in[13];
  const float* upd_w1      = (const float*)d_in[14];
  const float* upd_b1      = (const float*)d_in[15];
  const float* upd_w2      = (const float*)d_in[16];
  const float* upd_b2      = (const float*)d_in[17];
  const float* fc1_w       = (const float*)d_in[18];
  const float* fc1_b       = (const float*)d_in[19];
  const float* fc2_w       = (const float*)d_in[20];
  const float* fc2_b       = (const float*)d_in[21];
  const int*   from_idx    = (const int*)d_in[22];
  const int*   to_idx      = (const int*)d_in[23];
  const int*   graph_idx   = (const int*)d_in[24];
  const int*   qsizes      = (const int*)d_in[25];
  const int*   csizes      = (const int*)d_in[26];

  const int N = in_sizes[0] / 32;
  const int E = in_sizes[1] / 16;

  char* ws = (char*)d_ws;
  size_t off = 0;
  auto take = [&](size_t bytes) -> void* {
    void* p = ws + off;
    off = (off + bytes + 255) & ~(size_t)255;
    return p;
  };
  bf16*  wb     = (bf16*)take(365568 * sizeof(bf16));
  float* h      = (float*)take((size_t)N * 128 * 4);
  bf16*  hbf    = (bf16*)take((size_t)N * 128 * 2);
  bf16*  ebf    = (bf16*)take((size_t)E * 64 * 2);
  float* agg    = (float*)take((size_t)N * 128 * 4);
  int*   starts = (int*)take(1024 * 4);
  float* stk    = (float*)take((size_t)1024 * 16384 * 4);
  bf16*  stkbf  = (bf16*)take((size_t)1024 * 16384 * 2);
  bf16*  mqc    = (bf16*)take((size_t)1024 * 16384 * 2);
  float* la     = (float*)take((size_t)512 * 16384 * 4);
  bf16*  planbf = (bf16*)take((size_t)512 * 16384 * 2);

  bf16* w_encn = wb + 0;
  bf16* w_ence = wb + 4096;
  bf16* w_m1   = wb + 5120;
  bf16* w_m2   = wb + 87040;
  bf16* w_r1   = wb + 119808;
  bf16* w_r2   = wb + 201728;
  bf16* w_u1   = wb + 234496;
  bf16* w_u2   = wb + 300032;
  bf16* w_f1   = wb + 332800;
  bf16* w_f2   = wb + 349184;

  auto cv = [&](const float* s, bf16* d, int n) {
    k_f2bf<<<(n + 255) / 256, 256, 0, stream>>>(s, d, n);
  };
  cv(enc_node_w, w_encn, 4096);
  cv(enc_edge_w, w_ence, 1024);
  cv(msg_w1,  w_m1, 81920);
  cv(msg_w2,  w_m2, 32768);
  cv(rmsg_w1, w_r1, 81920);
  cv(rmsg_w2, w_r2, 32768);
  cv(upd_w1,  w_u1, 65536);
  cv(upd_w2,  w_u2, 32768);
  cv(fc1_w,   w_f1, 16384);
  cv(fc2_w,   w_f2, 16384);

  k_starts<<<1, 1024, 0, stream>>>(qsizes, csizes, starts);

  k_enc_nodes<<<(N + 15) / 16, 256, 0, stream>>>(node_f, w_encn, enc_node_b, h, hbf, N);
  k_enc_edges<<<(E + 15) / 16, 128, 0, stream>>>(edge_f, w_ence, enc_edge_b, ebf, E);

  for (int round = 0; round < 5; ++round) {
    (void)hipMemsetAsync(agg, 0, (size_t)N * 128 * 4, stream);
    k_msg<<<(E + 15) / 16, 256, 0, stream>>>(hbf, ebf, from_idx, to_idx,
                                             w_m1, msg_b1, w_m2, msg_b2,
                                             w_r1, rmsg_b1, w_r2, rmsg_b2, agg, E);
    k_upd<<<(N + 15) / 16, 256, 0, stream>>>(agg, h, hbf, w_u1, upd_b1, w_u2, upd_b2, N);
  }

  (void)hipMemsetAsync(stk, 0, (size_t)1024 * 16384 * 4, stream);
  (void)hipMemsetAsync(stkbf, 0, (size_t)1024 * 16384 * 2, stream);
  k_scatter<<<(N + 7) / 8, 256, 0, stream>>>(h, graph_idx, starts, stk, stkbf, N);

  k_tqc<<<1024, 256, 0, stream>>>(stkbf, w_f1, fc1_b, w_f2, fc2_b, qsizes, csizes, mqc);
  k_logalpha<<<512, 256, 0, stream>>>(mqc, la);
  k_sinkhorn<<<512, 256, 0, stream>>>(la, planbf);
  k_scores<<<512, 256, 0, stream>>>(planbf, stkbf, stk, (float*)d_out);

  (void)n_in; (void)out_size; (void)ws_size;
}
